// RoPEAttention_19207093747873
// MI455X (gfx1250) — matrix operand order repaired and checked by reading, not compiled, not run on hardware
//
#include <hip/hip_runtime.h>
#include <hip/hip_bf16.h>
#include <stdint.h>

// ---------------------------------------------------------------------------
// CDNA5 (gfx1250) wave32 WMMA attention block with async global->LDS staging.
//   qkv = f16(x) @ f16(c_attn_w)^T-staged + bias + pos       (WMMA f16->f32)
//   y   = flash-attention(q, k, v) per (b, head)             (WMMA f16->f32)
//   out = f16(y) @ f16(c_proj_w) + bias                      (WMMA f16->f32)
// Data movement: GLOBAL_LOAD_ASYNC_TO_LDS_B128 (ASYNCcnt), double-buffered.
// ---------------------------------------------------------------------------

typedef __attribute__((ext_vector_type(16))) _Float16 v16h;
typedef __attribute__((ext_vector_type(8)))  float    v8f;
typedef int v4i_ __attribute__((vector_size(16)));   // matches builtin param type

#define NHEAD 16
#define HS    64
#define TT    2048
#define CC    1024
#define BB    4
#define BT    (BB * TT)      // 8192 rows
#define C3    (3 * CC)       // 3072

union F16Frag { uint4 q[2]; v16h h; };

// ---------------------------------------------------------------------------
// Async global->LDS copy of 16 bytes (b128), tracked with ASYNCcnt.
// Signature (probe-confirmed by round-2 diagnostic):
//   void __builtin_amdgcn_global_load_async_to_lds_b128(v4i as(1)* gsrc,
//                                                       v4i as(3)* ldst,
//                                                       imm offset, imm cpol)
// ---------------------------------------------------------------------------
__device__ __forceinline__ void async_copy_b128(const _Float16* gsrc, _Float16* ldst) {
#if __has_builtin(__builtin_amdgcn_global_load_async_to_lds_b128)
    __builtin_amdgcn_global_load_async_to_lds_b128(
        (__attribute__((address_space(1))) v4i_*)gsrc,
        (__attribute__((address_space(3))) v4i_*)ldst, 0, 0);
#else
    uint32_t loff = (uint32_t)(uintptr_t)(__attribute__((address_space(3))) void*)ldst;
    asm volatile("global_load_async_to_lds_b128 %0, %1, off"
                 :: "v"(loff), "v"((uint64_t)(uintptr_t)gsrc) : "memory");
#endif
}

__device__ __forceinline__ void wait_async0() {
#if __has_builtin(__builtin_amdgcn_s_wait_asynccnt)
    __builtin_amdgcn_s_wait_asynccnt(0);
#else
    asm volatile("s_wait_asynccnt 0x0" ::: "memory");
#endif
}

// ---------------------------------------------------------------------------
// WMMA fragment helpers (CDNA5 ISA 7.12.2 wave32 16-bit layouts).
// ---------------------------------------------------------------------------
__device__ __forceinline__ v16h load_frag_a(const _Float16* base, int ldk, int lane) {
    int m = lane & 15;
    int hh = (lane >> 4) << 3;               // 8 * half
    const _Float16* p = base + m * ldk + hh;
    F16Frag f;
    f.q[0] = *(const uint4*)(p);             // halfs [8h, 8h+8)
    f.q[1] = *(const uint4*)(p + 16);        // halfs [16+8h, 16+8h+8)
    return f.h;
}

// B tile stored TRANSPOSED as [n][k], row stride ldk halfs.
__device__ __forceinline__ v16h load_frag_b(const _Float16* base, int ldk, int lane) {
    int n = lane & 15;
    int hh = (lane >> 4) << 3;               // 8 * half
    const _Float16* p = base + n * ldk + hh;
    F16Frag f;
    f.q[0] = *(const uint4*)(p);
    f.q[1] = *(const uint4*)(p + 16);
    return f.h;
}

__device__ __forceinline__ v8f wmma_f16(v16h a, v16h b, v8f c) {
    return __builtin_amdgcn_wmma_f32_16x16x32_f16(false, a, false, b,
                                                  (short)0, c, false, false);
}

// ---------------------------------------------------------------------------
// Staging kernels (run once, off the critical loop):
//   cvt_f16:   f32 -> f16 elementwise
//   cvt_t:     f32 [K][N] -> f16 [N][K]  (weights, so GEMM B tiles are linear)
//   vtrans:    qkv v-slab -> f16 [B][H][hs][T] (so PV B tiles are linear)
// ---------------------------------------------------------------------------
__global__ void __launch_bounds__(256)
cvt_f16_kernel(const float* __restrict__ src, _Float16* __restrict__ dst, int n) {
    int i = blockIdx.x * 256 + threadIdx.x;
    if (i < n) dst[i] = (_Float16)src[i];
}

__global__ void __launch_bounds__(256)
cvt_t_kernel(const float* __restrict__ src, _Float16* __restrict__ dst, int K, int N) {
    int i = blockIdx.x * 256 + threadIdx.x;  // dst index = n*K + k
    if (i < K * N) {
        int k = i % K;
        int n = i / K;
        dst[i] = (_Float16)src[(size_t)k * N + n];
    }
}

__global__ void __launch_bounds__(256)
vtrans_kernel(const _Float16* __restrict__ qkv, _Float16* __restrict__ vT) {
    int i = blockIdx.x * 256 + threadIdx.x;          // output index
    int t = i & (TT - 1);
    int s = (i >> 11) & (HS - 1);
    int h = (i >> 17) & (NHEAD - 1);
    int b = i >> 21;
    vT[i] = qkv[(size_t)(b * TT + t) * C3 + 2 * CC + h * HS + s];
}

// ---------------------------------------------------------------------------
// Tiled WMMA GEMM:  out[M,N] = A[M,K](f16) @ Bt[N,K](f16, pre-transposed) + bias
// mode 0: fused bias + positional add (reference's (B,NH,T,hs) reshape), f16 out
// mode 1: bias only, f32 out
// Tile 128x128, K-step 32, 256 threads = 8 wave32, async double-buffered LDS.
// ---------------------------------------------------------------------------
#define GMT 128
#define GNT 128
#define GKS 32

__device__ __forceinline__ void gemm_issue_tile(const _Float16* __restrict__ A,
                                                const _Float16* __restrict__ Bt,
                                                _Float16* As, _Float16* Bs,
                                                int K, int m0, int n0, int k0, int tid) {
    const int row = tid >> 1, seg = tid & 1;         // 128 rows x 2 segs of 16 halfs
    const _Float16* ga = A + (size_t)(m0 + row) * K + k0 + seg * 16;
    _Float16* la = As + row * GKS + seg * 16;
    async_copy_b128(ga, la);
    async_copy_b128(ga + 8, la + 8);
    const _Float16* gb = Bt + (size_t)(n0 + row) * K + k0 + seg * 16;
    _Float16* lb = Bs + row * GKS + seg * 16;
    async_copy_b128(gb, lb);
    async_copy_b128(gb + 8, lb + 8);
}

__global__ void __launch_bounds__(256)
gemm_wmma_kernel(const _Float16* __restrict__ A, const _Float16* __restrict__ Bt,
                 const float* __restrict__ bias, const float* __restrict__ wpe,
                 _Float16* __restrict__ outh, float* __restrict__ outf,
                 int M, int N, int K, int mode) {
    __shared__ __align__(16) _Float16 As[2][GMT * GKS];   // [m][k], 2 x 8 KB
    __shared__ __align__(16) _Float16 Bs[2][GNT * GKS];   // [n][k], 2 x 8 KB

    const int tid  = threadIdx.x;
    const int lane = tid & 31;
    const int wave = tid >> 5;
    const int wm   = wave >> 1;                        // 0..3
    const int wn   = wave & 1;                         // 0..1
    const int m0   = blockIdx.y * GMT;
    const int n0   = blockIdx.x * GNT;

    v8f acc[2][4];
#pragma unroll
    for (int i = 0; i < 2; ++i)
#pragma unroll
        for (int j = 0; j < 4; ++j) acc[i][j] = (v8f)0.0f;

    const int ntiles = K / GKS;
    gemm_issue_tile(A, Bt, As[0], Bs[0], K, m0, n0, 0, tid);

    for (int t = 0; t < ntiles; ++t) {
        const int cur = t & 1;
        wait_async0();                 // this wave's tile-t copies landed in LDS
        __syncthreads();               // all waves' copies landed; prior reads done
        if (t + 1 < ntiles)
            gemm_issue_tile(A, Bt, As[cur ^ 1], Bs[cur ^ 1], K, m0, n0,
                            (t + 1) * GKS, tid);

        v16h af[2], bf[4];
#pragma unroll
        for (int fm = 0; fm < 2; ++fm)
            af[fm] = load_frag_a(As[cur] + (wm * 32 + fm * 16) * GKS, GKS, lane);
#pragma unroll
        for (int fn = 0; fn < 4; ++fn)
            bf[fn] = load_frag_b(Bs[cur] + (wn * 64 + fn * 16) * GKS, GKS, lane);
#pragma unroll
        for (int fm = 0; fm < 2; ++fm)
#pragma unroll
            for (int fn = 0; fn < 4; ++fn)
                acc[fm][fn] = wmma_f16(af[fm], bf[fn], acc[fm][fn]);
    }

    // Epilogue. C/D layout: VGPR r -> row (r + 8*half), col = lane&15.
    const int nlo  = lane & 15;
    const int half = lane >> 4;
#pragma unroll
    for (int fm = 0; fm < 2; ++fm) {
#pragma unroll
        for (int fn = 0; fn < 4; ++fn) {
            const int col = n0 + wn * 64 + fn * 16 + nlo;
            const float bcol = bias[col];
#pragma unroll
            for (int r = 0; r < 8; ++r) {
                const int row = m0 + wm * 32 + fm * 16 + r + 8 * half;
                float v = acc[fm][fn][r] + bcol;
                if (mode == 0) {
                    if (col < 2 * CC) {
                        // reference does q.reshape(B, NH, T, hs) + wpe[:T]
                        const int jj   = col & (CC - 1);
                        const int trow = row & (TT - 1);
                        const int f64  = trow * (CC / HS) + (jj >> 6);
                        const int tt   = f64 & (TT - 1);
                        const int ss   = col & (HS - 1);
                        v += wpe[tt * HS + ss];
                    }
                    outh[(size_t)row * N + col] = (_Float16)v;
                } else {
                    outf[(size_t)row * N + col] = v;
                }
            }
        }
    }
}

// ---------------------------------------------------------------------------
// Flash attention: one workgroup per (q-block of 128, b*NHEAD + h).
// 8 waves; wave w owns q rows [w*16, w*16+16). Key/value blocks of 64,
// async double-buffered in LDS.
// ---------------------------------------------------------------------------
#define QB 128
#define KB 64

__device__ __forceinline__ void attn_issue_kv(const _Float16* __restrict__ qkv,
                                              const _Float16* __restrict__ vT,
                                              _Float16* Ks, _Float16* Vs,
                                              int b, int h, int jb, int tid) {
    const int r = tid >> 2, seg = tid & 3;           // 64 rows x 4 segs of 16 halfs
    const _Float16* gk = qkv + (size_t)(b * TT + jb * KB + r) * C3 + CC + h * HS + seg * 16;
    _Float16* lk = Ks + r * HS + seg * 16;
    async_copy_b128(gk, lk);
    async_copy_b128(gk + 8, lk + 8);
    const _Float16* gv = vT + ((size_t)((b * NHEAD + h) * HS + r)) * TT + jb * KB + seg * 16;
    _Float16* lv = Vs + r * KB + seg * 16;
    async_copy_b128(gv, lv);
    async_copy_b128(gv + 8, lv + 8);
}

__global__ void __launch_bounds__(256)
attn_wmma_kernel(const _Float16* __restrict__ qkv, const _Float16* __restrict__ vT,
                 _Float16* __restrict__ yh) {
    __shared__ __align__(16) _Float16 Qs[QB * HS];         // [qrow][s]      16 KB
    __shared__ __align__(16) _Float16 Ks[2][KB * HS];      // [key][s]   2 x  8 KB
    __shared__ __align__(16) _Float16 Vst[2][HS * KB];     // [s][key]   2 x  8 KB
    __shared__ __align__(16) _Float16 Ps[8 * 16 * KB];     // per-wave P     16 KB

    const int tid  = threadIdx.x;
    const int lane = tid & 31;
    const int wave = tid >> 5;
    const int nlo  = lane & 15;
    const int half = lane >> 4;

    const int qb = blockIdx.x;
    const int b  = blockIdx.y >> 4;
    const int h  = blockIdx.y & 15;
    const int qbase = qb * QB;

    // stage Q block: 128 rows x 64 halfs, async
    {
        const int r = tid >> 1, seg = tid & 1;
        const _Float16* gq = qkv + (size_t)(b * TT + qbase + r) * C3 + h * HS + seg * 32;
        _Float16* lq = Qs + r * HS + seg * 32;
        async_copy_b128(gq, lq);
        async_copy_b128(gq + 8,  lq + 8);
        async_copy_b128(gq + 16, lq + 16);
        async_copy_b128(gq + 24, lq + 24);
    }

    float mrow[8], lrow[8];
    v8f accO[4];
#pragma unroll
    for (int r = 0; r < 8; ++r) { mrow[r] = -1.0e30f; lrow[r] = 0.0f; }
#pragma unroll
    for (int fo = 0; fo < 4; ++fo) accO[fo] = (v8f)0.0f;

    const float scale = 0.125f;          // 1/sqrt(64)
    const float L2E = 1.44269504f;
    const int jbmax = 2 * qb + 1;        // last key block overlapping q rows

    attn_issue_kv(qkv, vT, Ks[0], Vst[0], b, h, 0, tid);

    for (int jb = 0; jb <= jbmax; ++jb) {
        const int cur = jb & 1;
        wait_async0();                   // Q (first iter) + K/V block jb landed
        __syncthreads();
        if (jb + 1 <= jbmax)
            attn_issue_kv(qkv, vT, Ks[cur ^ 1], Vst[cur ^ 1], b, h, jb + 1, tid);

        // S = Q @ K^T   (16 q-rows x 64 keys per wave)
        v8f sf[4];
#pragma unroll
        for (int fn = 0; fn < 4; ++fn) sf[fn] = (v8f)0.0f;
#pragma unroll
        for (int ks = 0; ks < 2; ++ks) {
            v16h aq = load_frag_a(Qs + (wave * 16) * HS + ks * 32, HS, lane);
#pragma unroll
            for (int fn = 0; fn < 4; ++fn)
                sf[fn] = wmma_f16(aq, load_frag_b(Ks[cur] + (fn * 16) * HS + ks * 32, HS, lane), sf[fn]);
        }

        // online softmax (rows live in 16-lane half-waves: row = r + 8*half)
#pragma unroll
        for (int r = 0; r < 8; ++r) {
            const int qr = qbase + wave * 16 + r + 8 * half;
            float mx = -3.0e38f;
#pragma unroll
            for (int fn = 0; fn < 4; ++fn) {
                float s = sf[fn][r] * scale;
                const int key = jb * KB + fn * 16 + nlo;
                if (key > qr) s = -3.0e38f;         // causal mask
                sf[fn][r] = s;
                mx = fmaxf(mx, s);
            }
#pragma unroll
            for (int off = 8; off > 0; off >>= 1)
                mx = fmaxf(mx, __shfl_xor(mx, off, 16));
            const float newm = fmaxf(mrow[r], mx);
            const float alpha = exp2f((mrow[r] - newm) * L2E);
            float psum = 0.0f;
#pragma unroll
            for (int fn = 0; fn < 4; ++fn) {
                const float p = exp2f((sf[fn][r] - newm) * L2E);
                Ps[wave * (16 * KB) + (r + 8 * half) * KB + fn * 16 + nlo] = (_Float16)p;
                psum += p;
            }
#pragma unroll
            for (int off = 8; off > 0; off >>= 1)
                psum += __shfl_xor(psum, off, 16);
            lrow[r] = lrow[r] * alpha + psum;
            mrow[r] = newm;
#pragma unroll
            for (int fo = 0; fo < 4; ++fo) accO[fo][r] *= alpha;
        }

        // O += P @ V  (same-wave LDS writes/reads are in program order)
#pragma unroll
        for (int ks = 0; ks < 2; ++ks) {
            v16h ap = load_frag_a(Ps + wave * (16 * KB) + ks * 32, KB, lane);
#pragma unroll
            for (int fo = 0; fo < 4; ++fo)
                accO[fo] = wmma_f16(ap, load_frag_b(Vst[cur] + (fo * 16) * KB + ks * 32, KB, lane), accO[fo]);
        }
    }

    // normalize and store y as [B*T, C] f16
#pragma unroll
    for (int fo = 0; fo < 4; ++fo) {
#pragma unroll
        for (int r = 0; r < 8; ++r) {
            const float inv = 1.0f / lrow[r];
            const int trow = qbase + wave * 16 + r + 8 * half;
            const int col = h * HS + fo * 16 + nlo;
            yh[(size_t)(b * TT + trow) * CC + col] = (_Float16)(accO[fo][r] * inv);
        }
    }
}

// ---------------------------------------------------------------------------
extern "C" void kernel_launch(void* const* d_in, const int* in_sizes, int n_in,
                              void* d_out, int out_size, void* d_ws, size_t ws_size,
                              hipStream_t stream) {
    const float* x   = (const float*)d_in[0];   // [B*T, C]
    const float* caw = (const float*)d_in[1];   // [C, 3C]
    const float* cab = (const float*)d_in[2];   // [3C]
    const float* cpw = (const float*)d_in[3];   // [C, C]
    const float* cpb = (const float*)d_in[4];   // [C]
    const float* wpe = (const float*)d_in[5];   // [T, hs]
    float* out = (float*)d_out;                 // [B*T, C]

    _Float16* ws = (_Float16*)d_ws;
    size_t o = 0;
    _Float16* xh   = ws + o; o += (size_t)BT * CC;     // x,  f16
    _Float16* wahT = ws + o; o += (size_t)CC * C3;     // c_attn_w^T [3C][C], f16
    _Float16* wphT = ws + o; o += (size_t)CC * CC;     // c_proj_w^T [C][C],  f16
    _Float16* qkvh = ws + o; o += (size_t)BT * C3;     // qkv, f16
    _Float16* vT   = ws + o; o += (size_t)BT * CC;     // v as [B][H][hs][T], f16
    _Float16* yh   = ws + o;                           // attention out, f16

    const int nx = BT * CC, nw1 = CC * C3, nw2 = CC * CC, nv = BT * CC;
    cvt_f16_kernel<<<(nx  + 255) / 256, 256, 0, stream>>>(x, xh, nx);
    cvt_t_kernel<<<(nw1 + 255) / 256, 256, 0, stream>>>(caw, wahT, CC, C3);
    cvt_t_kernel<<<(nw2 + 255) / 256, 256, 0, stream>>>(cpw, wphT, CC, CC);

    // qkv = x @ c_attn_w + bias + pos   (f16 out)
    gemm_wmma_kernel<<<dim3(C3 / GNT, BT / GMT), 256, 0, stream>>>(
        xh, wahT, cab, wpe, qkvh, nullptr, BT, C3, CC, 0);

    // v slab -> [B][H][hs][T]
    vtrans_kernel<<<nv / 256, 256, 0, stream>>>(qkvh, vT);

    // y = attention(q, k, v)
    attn_wmma_kernel<<<dim3(TT / QB, BB * NHEAD), 256, 0, stream>>>(qkvh, vT, yh);

    // out = y @ c_proj_w + bias   (f32 out)
    gemm_wmma_kernel<<<dim3(CC / GNT, BT / GMT), 256, 0, stream>>>(
        yh, wphT, cpb, nullptr, nullptr, out, BT, CC, CC, 1);
}
